// WSDModel_13872744366164
// MI455X (gfx1250) — matrix operand order repaired and checked by reading, not compiled, not run on hardware
//
#include <hip/hip_runtime.h>

#define DEVINL __device__ __forceinline__

typedef __attribute__((ext_vector_type(16))) __bf16 bf16x16;
typedef __attribute__((ext_vector_type(8)))  __bf16 bf16x8;
typedef __attribute__((ext_vector_type(8)))  float  f32x8;
typedef __attribute__((__vector_size__(16))) int    v4i;

enum { EPI_BF16 = 0, EPI_LOGITS = 1, EPI_F32 = 2, EPI_RESID = 3 };

#if defined(__HIP_DEVICE_COMPILE__) && defined(__gfx1250__) && \
    __has_builtin(__builtin_amdgcn_global_load_async_to_lds_b128)
#define USE_ASYNC_LDS 1
#else
#define USE_ASYNC_LDS 0
#endif

DEVINL unsigned short f2bf(float f) {
    unsigned u = __float_as_uint(f);
    u += 0x7FFFu + ((u >> 16) & 1u);            // round-to-nearest-even
    return (unsigned short)(u >> 16);
}

#if USE_ASYNC_LDS
// one 16-byte async DMA: global -> LDS (tracked by ASYNCcnt)
DEVINL void async_ld_b128(const unsigned short* g, unsigned short* l) {
    __builtin_amdgcn_global_load_async_to_lds_b128(
        (__attribute__((address_space(1))) v4i*)(g),
        (__attribute__((address_space(3))) v4i*)(l), 0, 0);
}
DEVINL void wait_async0() {
#if __has_builtin(__builtin_amdgcn_s_wait_asynccnt)
    __builtin_amdgcn_s_wait_asynccnt(0);
#else
    asm volatile("s_wait_asynccnt 0" ::: "memory");
#endif
}
#endif

// ---------------------------------------------------------------------------
// bf16 WMMA GEMM: C[M,N] = A[M,K] * B^T, with B supplied as [N,K] row-major
// (every producer lays B out transposed so ALL LDS staging is contiguous
// b128 traffic -- no 16-bit scatter).  128x128 block tile, K-step 32,
// 8 waves in 2x4 layout, each wave = 4x2 v_wmma_f32_16x16x32_bf16 fragments.
// Tile staging: async DMA to LDS (ASYNCcnt, double-buffered) when available,
// else a 2-stage register pipeline hiding global latency behind the WMMAs.
//   TRI_SKIP : skip block tiles strictly above the causal diagonal
//   TRI_K    : clamp K loop at m0+128 (A rows are causally lower-triangular)
// ---------------------------------------------------------------------------
template<int EPI, bool TRI_SKIP, bool TRI_K>
__global__ __launch_bounds__(256)
void gemm_wmma(const unsigned short* __restrict__ Am, long long sAb, int lda,
               const unsigned short* __restrict__ Bm, long long sBb, int ldb,
               void* __restrict__ Cm, long long sCb, int ldc,
               int M, int N, int K,
               const float* __restrict__ resid, int ldr,
               const int* __restrict__ tokens, int tokStride)
{
#if USE_ASYNC_LDS
    __shared__ __attribute__((aligned(32))) unsigned short lA[2][128 * 32]; // [m][k]
    __shared__ __attribute__((aligned(32))) unsigned short lB[2][128 * 32]; // [n][k]
#else
    __shared__ __attribute__((aligned(32))) unsigned short lA[1][128 * 32];
    __shared__ __attribute__((aligned(32))) unsigned short lB[1][128 * 32];
#endif

    const int t    = threadIdx.x;
    const int wave = t >> 5;
    const int lane = t & 31;
    const int m0   = blockIdx.y * 128;
    const int n0   = blockIdx.x * 128;
    const int bz   = blockIdx.z;

    if (TRI_SKIP && n0 > m0 + 127) return;      // entire tile is future -> skipped

    const unsigned short* Ab = Am + (long long)bz * sAb;
    const unsigned short* Bb = Bm + (long long)bz * sBb;

    const int wm = (wave >> 2) * 64;            // wave row offset within tile
    const int wn = (wave & 3) * 32;             // wave col offset within tile

    f32x8 acc[4][2];
#pragma unroll
    for (int mi = 0; mi < 4; ++mi)
#pragma unroll
        for (int ni = 0; ni < 2; ++ni)
            acc[mi][ni] = (f32x8){0.f, 0.f, 0.f, 0.f, 0.f, 0.f, 0.f, 0.f};

    int Keff = K;
    if (TRI_K) { int kl = m0 + 128; Keff = (kl < K) ? kl : K; }

    // staging coordinates: each thread owns 16 bf16 (32 B) of each tile
    const int srow = t >> 1;
    const int scb  = (t & 1) * 16;
    const unsigned short* gA = Ab + (long long)(m0 + srow) * lda + scb;
    const unsigned short* gB = Bb + (long long)(n0 + srow) * ldb + scb;
    const int ldsoff = srow * 32 + scb;

    const int arow = lane & 15;
    const int kh   = (lane >> 4) & 1;

#if USE_ASYNC_LDS
    // ---------------- async DMA double-buffered staging --------------------
    auto issue = [&](int kt, int bufi) {
        async_ld_b128(gA + kt,     &lA[bufi][ldsoff]);
        async_ld_b128(gA + kt + 8, &lA[bufi][ldsoff + 8]);
        async_ld_b128(gB + kt,     &lB[bufi][ldsoff]);
        async_ld_b128(gB + kt + 8, &lB[bufi][ldsoff + 8]);
    };
    int cur = 0;
    issue(0, 0);
    for (int kt = 0; kt < Keff; kt += 32) {
        wait_async0();                 // this wave's DMA into buf[cur] landed
        __syncthreads();               // everyone's DMA landed
        if (kt + 32 < Keff) issue(kt + 32, cur ^ 1);
        const unsigned short* pA = lA[cur];
        const unsigned short* pB = lB[cur];
#else
    // ---------------- register-pipelined staging ---------------------------
    uint4 ra0, ra1, rb0, rb1;
    auto loadg = [&](int kt) {
        ra0 = ((const uint4*)(gA + kt))[0];
        ra1 = ((const uint4*)(gA + kt))[1];
        rb0 = ((const uint4*)(gB + kt))[0];
        rb1 = ((const uint4*)(gB + kt))[1];
        __builtin_prefetch(gA + kt + 32, 0, 1);     // global_prefetch_b8
        __builtin_prefetch(gB + kt + 32, 0, 1);
    };
    loadg(0);
    for (int kt = 0; kt < Keff; kt += 32) {
        __syncthreads();               // previous iteration's readers are done
        *(uint4*)&lA[0][ldsoff]     = ra0;
        *(uint4*)&lA[0][ldsoff + 8] = ra1;
        *(uint4*)&lB[0][ldsoff]     = rb0;
        *(uint4*)&lB[0][ldsoff + 8] = rb1;
        __syncthreads();               // tile visible to all waves
        if (kt + 32 < Keff) loadg(kt + 32);   // overlap next loads with WMMAs
        const unsigned short* pA = lA[0];
        const unsigned short* pB = lB[0];
#endif
        // ---- build fragments, contiguous K order k = kh*16 + s in A and B -------------------------
        bf16x16 afr[4], bfr[2];
#pragma unroll
        for (int mi = 0; mi < 4; ++mi) {
            const unsigned short* p = &pA[(wm + mi * 16 + arow) * 32];
            bf16x8 lo = *(const bf16x8*)(p + kh * 16);       // K = kh*16 .. +7
            bf16x8 hi = *(const bf16x8*)(p + kh * 16 + 8);   // K = kh*16+8 .. +15
            afr[mi] = __builtin_shufflevector(lo, hi, 0, 1, 2, 3, 4, 5, 6, 7,
                                              8, 9, 10, 11, 12, 13, 14, 15);
        }
#pragma unroll
        for (int ni = 0; ni < 2; ++ni) {
            const unsigned short* p = &pB[(wn + ni * 16 + arow) * 32 + kh * 16];
            bfr[ni] = *(const bf16x16*)p;                    // 16 contiguous K
        }
        // ---- 8 WMMAs per wave per K-step ----------------------------------
#pragma unroll
        for (int mi = 0; mi < 4; ++mi)
#pragma unroll
            for (int ni = 0; ni < 2; ++ni)
                acc[mi][ni] = __builtin_amdgcn_wmma_f32_16x16x32_bf16(
                    false, afr[mi], false, bfr[ni],
                    (short)0, acc[mi][ni], false, false);
#if USE_ASYNC_LDS
        cur ^= 1;
    }
#else
    }
#endif

    // ---- epilogue: C/D layout is M = 8*(lane>=16)+r, N = lane%16 ----------
    const int cn = lane & 15;
    const int mh = (lane >> 4) & 1;
#pragma unroll
    for (int mi = 0; mi < 4; ++mi) {
#pragma unroll
        for (int ni = 0; ni < 2; ++ni) {
#pragma unroll
            for (int r = 0; r < 8; ++r) {
                int gm = m0 + wm + mi * 16 + mh * 8 + r;
                int gn = n0 + wn + ni * 16 + cn;
                float v = acc[mi][ni][r];
                if (EPI == EPI_BF16) {
                    unsigned short* C = (unsigned short*)Cm + (long long)bz * sCb;
                    C[(long long)gm * ldc + gn] = f2bf(v);
                } else if (EPI == EPI_F32) {
                    float* C = (float*)Cm + (long long)bz * sCb;
                    C[(long long)gm * ldc + gn] = v;
                } else if (EPI == EPI_RESID) {
                    float* C = (float*)Cm + (long long)bz * sCb;
                    C[(long long)gm * ldc + gn] = v + resid[(long long)gm * ldr + gn];
                } else { // EPI_LOGITS: causal/positional bias + padding mask
                    float* C = (float*)Cm + (long long)bz * sCb;
                    float out;
                    if (gn > gm || tokens[(long long)bz * tokStride + gn] == 0)
                        out = -__builtin_inff();
                    else
                        out = v - (float)(gm - gn);
                    C[(long long)gm * ldc + gn] = out;
                }
            }
        }
    }
}

// ---------------------------------------------------------------------------
// Elementwise f32 -> bf16 cast
// ---------------------------------------------------------------------------
__global__ __launch_bounds__(256)
void cast_f32_bf16(const float* __restrict__ src, unsigned short* __restrict__ dst,
                   long long n)
{
    long long idx    = (long long)blockIdx.x * blockDim.x + threadIdx.x;
    long long stride = (long long)gridDim.x * blockDim.x;
    for (; idx < n; idx += stride) dst[idx] = f2bf(src[idx]);
}

// ---------------------------------------------------------------------------
// Fused cast+transpose: dst[c][r] (bf16, ld=R) = src[r][c] (f32, ld=C)
// 32x32 LDS tile, 256 threads, coalesced on both sides.
// ---------------------------------------------------------------------------
__global__ __launch_bounds__(256)
void transpose_cast_f32_bf16(const float* __restrict__ src,
                             unsigned short* __restrict__ dst, int R, int C)
{
    __shared__ unsigned short tile[32][33];
    const int r0 = blockIdx.y * 32, c0 = blockIdx.x * 32;
    const int tx = threadIdx.x & 31, ty = threadIdx.x >> 5;
#pragma unroll
    for (int i = 0; i < 32; i += 8)
        tile[ty + i][tx] = f2bf(src[(long long)(r0 + ty + i) * C + c0 + tx]);
    __syncthreads();
#pragma unroll
    for (int i = 0; i < 32; i += 8)
        dst[(long long)(c0 + ty + i) * R + r0 + tx] = tile[tx][ty + i];
}

// ---------------------------------------------------------------------------
// bf16 transpose (batched via blockIdx.z): dst[c][r] = src[r][c]
// ---------------------------------------------------------------------------
__global__ __launch_bounds__(256)
void transpose_bf16(const unsigned short* __restrict__ srcb,
                    unsigned short* __restrict__ dstb, int R, int C)
{
    __shared__ unsigned short tile[32][33];
    const long long boff = (long long)blockIdx.z * R * C;
    const unsigned short* src = srcb + boff;
    unsigned short*       dst = dstb + boff;
    const int r0 = blockIdx.y * 32, c0 = blockIdx.x * 32;
    const int tx = threadIdx.x & 31, ty = threadIdx.x >> 5;
#pragma unroll
    for (int i = 0; i < 32; i += 8)
        tile[ty + i][tx] = src[(long long)(r0 + ty + i) * C + c0 + tx];
    __syncthreads();
#pragma unroll
    for (int i = 0; i < 32; i += 8)
        dst[(long long)(c0 + ty + i) * R + r0 + tx] = tile[tx][ty + i];
}

// ---------------------------------------------------------------------------
// X = E_v[M_s] gather; writes fp32 (residual path) and bf16 (WMMA path)
// ---------------------------------------------------------------------------
__global__ __launch_bounds__(128)
void gather_embed(const int* __restrict__ M_s, const float* __restrict__ E_v,
                  float* __restrict__ Xf, unsigned short* __restrict__ Xh, int Dv)
{
    long long row = blockIdx.x;                 // b*N + n
    int tok = M_s[row];
    const float* src = E_v + (long long)tok * Dv;
    float* df          = Xf + row * Dv;
    unsigned short* dh = Xh + row * Dv;
    for (int c = threadIdx.x; c < Dv; c += 128) {
        float v = src[c];
        df[c] = v;
        dh[c] = f2bf(v);
    }
}

// ---------------------------------------------------------------------------
// Causal softmax over rows of length Nn; j>i entries written as exact 0
// (their logits are never read).  One 256-thread block (8 waves) per row.
// ---------------------------------------------------------------------------
__global__ __launch_bounds__(256)
void softmax_rows(float* __restrict__ Aout, int Nn)
{
    const long long row = blockIdx.x;
    const int i = (int)(row % Nn);
    float* p = Aout + row * (long long)Nn;
    const int t = threadIdx.x;
    __shared__ float redm[8];
    __shared__ float reds[8];

    float mx = -__builtin_inff();
    for (int j = t; j <= i; j += 256) mx = fmaxf(mx, p[j]);
#pragma unroll
    for (int off = 16; off; off >>= 1) mx = fmaxf(mx, __shfl_xor(mx, off, 32));
    if ((t & 31) == 0) redm[t >> 5] = mx;
    __syncthreads();
    if (t == 0) {
        float m = redm[0];
        for (int w = 1; w < 8; ++w) m = fmaxf(m, redm[w]);
        redm[0] = m;
    }
    __syncthreads();
    mx = redm[0];

    float s = 0.f;
    for (int j = t; j <= i; j += 256) s += __expf(p[j] - mx);
#pragma unroll
    for (int off = 16; off; off >>= 1) s += __shfl_xor(s, off, 32);
    if ((t & 31) == 0) reds[t >> 5] = s;
    __syncthreads();
    if (t == 0) {
        float a = 0.f;
        for (int w = 0; w < 8; ++w) a += reds[w];
        reds[0] = a;
    }
    __syncthreads();
    const float inv = 1.0f / reds[0];

    for (int j = t; j < Nn; j += 256)
        p[j] = (j <= i) ? __expf(p[j] - mx) * inv : 0.0f;
}

// ---------------------------------------------------------------------------
// LayerNorm over D=512; writes bf16 H for the final WMMA GEMM.
// ---------------------------------------------------------------------------
__global__ __launch_bounds__(256)
void layernorm_bf16(const float* __restrict__ Xin, const float* __restrict__ gamma,
                    const float* __restrict__ beta, unsigned short* __restrict__ Hh,
                    int Dv)
{
    const long long row = blockIdx.x;
    const float* x = Xin + row * Dv;
    const int t = threadIdx.x;
    __shared__ float rs[8];
    __shared__ float rq[8];

    float s = 0.f, q = 0.f;
    for (int c = t; c < Dv; c += 256) { float v = x[c]; s += v; q += v * v; }
#pragma unroll
    for (int off = 16; off; off >>= 1) { s += __shfl_xor(s, off, 32); q += __shfl_xor(q, off, 32); }
    if ((t & 31) == 0) { rs[t >> 5] = s; rq[t >> 5] = q; }
    __syncthreads();
    if (t == 0) {
        float a = 0.f, b = 0.f;
        for (int w = 0; w < 8; ++w) { a += rs[w]; b += rq[w]; }
        rs[0] = a; rq[0] = b;
    }
    __syncthreads();
    const float mu  = rs[0] / (float)Dv;
    const float var = rq[0] / (float)Dv - mu * mu;
    const float inv = rsqrtf(var + 1e-5f);

    unsigned short* out = Hh + row * Dv;
    for (int c = t; c < Dv; c += 256)
        out[c] = f2bf((x[c] - mu) * inv * gamma[c] + beta[c]);
}

// ---------------------------------------------------------------------------
// Host side
// ---------------------------------------------------------------------------
extern "C" void kernel_launch(void* const* d_in, const int* in_sizes, int n_in,
                              void* d_out, int out_size, void* d_ws, size_t ws_size,
                              hipStream_t stream)
{
    (void)in_sizes; (void)n_in; (void)out_size; (void)ws_size;
    const int Bb = 4, Nn = 4096, Dd = 512, Yy = 16000;

    const int*   M_s   = (const int*)d_in[0];
    const float* E_v   = (const float*)d_in[1];
    const float* E_y   = (const float*)d_in[2];
    const float* W_A   = (const float*)d_in[3];
    const float* W_O   = (const float*)d_in[4];
    const float* gamma = (const float*)d_in[5];
    const float* beta  = (const float*)d_in[6];

    float* y_out = (float*)d_out;                              // [B,N,Y]
    float* A_out = y_out + (long long)Bb * Nn * Yy;            // [B,N,N]

    char* w = (char*)d_ws;
    auto alloc = [&](long long bytes) -> char* {
        char* p = w; w += (bytes + 255) & ~255LL; return p;
    };
    float*          Xf  = (float*)alloc((long long)Bb * Nn * Dd * 4);          // X fp32
    unsigned short* Xh  = (unsigned short*)alloc((long long)Bb * Nn * Dd * 2); // X bf16 [N,D]
    unsigned short* Xt  = (unsigned short*)alloc((long long)Bb * Nn * Dd * 2); // X^T bf16 [D,N]
    unsigned short* XAh = (unsigned short*)alloc((long long)Bb * Nn * Dd * 2); // X*W_A bf16
    unsigned short* Ah  = (unsigned short*)alloc((long long)Bb * Nn * Nn * 2); // softmax(A) bf16
    unsigned short* AXh = (unsigned short*)alloc((long long)Bb * Nn * Dd * 2); // A*X bf16
    float*          Qcf = (float*)alloc((long long)Bb * Nn * Dd * 4);          // A*X*W_O + X
    unsigned short* Hh  = (unsigned short*)alloc((long long)Bb * Nn * Dd * 2); // LN(H) bf16
    unsigned short* WAt = (unsigned short*)alloc((long long)Dd * Dd * 2);      // W_A^T bf16
    unsigned short* WOt = (unsigned short*)alloc((long long)Dd * Dd * 2);      // W_O^T bf16
    unsigned short* EYh = (unsigned short*)alloc((long long)Yy * Dd * 2);      // E_y bf16 [Y,D]

    // 1) weights: cast (+transpose so GEMMs consume [N,K] row-major B)
    transpose_cast_f32_bf16<<<dim3(Dd / 32, Dd / 32), 256, 0, stream>>>(W_A, WAt, Dd, Dd);
    transpose_cast_f32_bf16<<<dim3(Dd / 32, Dd / 32), 256, 0, stream>>>(W_O, WOt, Dd, Dd);
    cast_f32_bf16<<<1024, 256, 0, stream>>>(E_y, EYh, (long long)Yy * Dd);

    // 2) X = E_v[M_s]; then X^T per batch for the A@X GEMM
    gather_embed<<<Bb * Nn, 128, 0, stream>>>(M_s, E_v, Xf, Xh, Dd);
    transpose_bf16<<<dim3(Dd / 32, Nn / 32, Bb), 256, 0, stream>>>(Xh, Xt, Nn, Dd);

    // 3) XA = X @ W_A            [16384,512] x [512,512] -> bf16
    gemm_wmma<EPI_BF16, false, false>
        <<<dim3(Dd / 128, (Bb * Nn) / 128, 1), 256, 0, stream>>>(
        Xh, 0, Dd, WAt, 0, Dd, XAh, 0, Dd, Bb * Nn, Dd, Dd, nullptr, 0, nullptr, 0);

    // 4) logits = XA @ X^T + bias (batched, future blocks skipped) -> fp32 in A_out
    gemm_wmma<EPI_LOGITS, true, false>
        <<<dim3(Nn / 128, Nn / 128, Bb), 256, 0, stream>>>(
        XAh, (long long)Nn * Dd, Dd, Xh, (long long)Nn * Dd, Dd,
        A_out, (long long)Nn * Nn, Nn, Nn, Nn, Dd, nullptr, 0, M_s, Nn);

    // 5) causal softmax in place (j>i written as 0)
    softmax_rows<<<Bb * Nn, 256, 0, stream>>>(A_out, Nn);

    // 6) A -> bf16
    cast_f32_bf16<<<4096, 256, 0, stream>>>(A_out, Ah, (long long)Bb * Nn * Nn);

    // 7) AX = A @ X (B given as X^T [D,N]; K loop clamped by causal structure)
    gemm_wmma<EPI_BF16, false, true>
        <<<dim3(Dd / 128, Nn / 128, Bb), 256, 0, stream>>>(
        Ah, (long long)Nn * Nn, Nn, Xt, (long long)Nn * Dd, Nn,
        AXh, (long long)Nn * Dd, Dd, Nn, Dd, Nn, nullptr, 0, nullptr, 0);

    // 8) Qc = AX @ W_O + X (residual fused) -> fp32
    gemm_wmma<EPI_RESID, false, false>
        <<<dim3(Dd / 128, (Bb * Nn) / 128, 1), 256, 0, stream>>>(
        AXh, 0, Dd, WOt, 0, Dd, Qcf, 0, Dd, Bb * Nn, Dd, Dd, Xf, Dd, nullptr, 0);

    // 9) H = LayerNorm(Qc) -> bf16
    layernorm_bf16<<<Bb * Nn, 256, 0, stream>>>(Qcf, gamma, beta, Hh, Dd);

    // 10) y_logits = H @ E_y^T   [16384,512] x [16000,512]^T -> fp32 out
    gemm_wmma<EPI_F32, false, false>
        <<<dim3(Yy / 128, (Bb * Nn) / 128, 1), 256, 0, stream>>>(
        Hh, 0, Dd, EYh, 0, Dd, y_out, 0, Yy, Bb * Nn, Yy, Dd, nullptr, 0, nullptr, 0);
}
